// Multihead_self_attention_60120952209649
// MI455X (gfx1250) — matrix operand order repaired and checked by reading, not compiled, not run on hardware
//
#include <hip/hip_runtime.h>
#include <stdint.h>

typedef __attribute__((ext_vector_type(16))) __bf16 v16bf;
typedef __attribute__((ext_vector_type(8)))  __bf16 v8bf;
typedef __attribute__((ext_vector_type(8)))  float  v8f;
typedef __attribute__((ext_vector_type(4)))  unsigned int u32x4;
typedef __attribute__((ext_vector_type(8)))  int i32x8;
typedef __attribute__((ext_vector_type(4)))  int i32x4;

#if defined(__AMDGCN__) && __has_builtin(__builtin_amdgcn_tensor_load_to_lds)
#define USE_TDM 1
#else
#define USE_TDM 0
#endif

union BF16x16 { v16bf v; v8bf h[2]; };

__device__ __forceinline__ v8f zero8() {
  v8f z = {0.f,0.f,0.f,0.f,0.f,0.f,0.f,0.f};
  return z;
}

__device__ __forceinline__ v8f wmma_bf16(v16bf a, v16bf b, v8f c) {
  // D = A(16x32 bf16) * B(32x16 bf16) + C(16x16 f32)
  return __builtin_amdgcn_wmma_f32_16x16x32_bf16(false, a, false, b, (short)0, c, false, false);
}

// ---------------------------------------------------------------------------
// TDM: DMA a 2-D tile of 2-byte elements (tile_w x tile_h, row stride
// `stride` elements) from global into LDS at byte offset lds_addr, inserting
// `pad_dw` DWORDs of LDS padding every (2<<pad_int_code) DWORDs streamed.
// D# layout per CDNA5 ISA (group2/3 zero: 2-D).
// This toolchain uses the 6-arg builtin (g0, g1, g2, g3, extra, cpol).
// ---------------------------------------------------------------------------
__device__ __forceinline__ void tdm_load_2d(unsigned lds_addr, const void* gptr,
                                            unsigned tile_w, unsigned tile_h,
                                            unsigned stride,
                                            unsigned pad_int_code, unsigned pad_dw) {
#if USE_TDM
  unsigned long long ga = (unsigned long long)(uintptr_t)gptr;
  u32x4 g0;
  g0[0] = 1u;                                            // count=1, user mode
  g0[1] = lds_addr;                                      // LDS byte address
  g0[2] = (unsigned)ga;                                  // global_addr[31:0]
  g0[3] = (unsigned)((ga >> 32) & 0x1FFFFFFu) | (2u << 30); // addr[56:32], type=2
  i32x8 g1;
  g1[0] = (int)((1u << 16)                               // data_size = 2 bytes
              | (1u << 20)                               // pad_enable
              | (pad_int_code << 22)                     // pad interval
              | ((pad_dw - 1u) << 25));                  // pad amount (DWORDs-1)
  g1[1] = (int)((tile_w & 0xFFFFu) << 16);               // tensor_dim0[15:0]
  g1[2] = (int)((tile_w >> 16) & 0xFFFFu)                // tensor_dim0[31:16]
        | (int)((tile_h & 0xFFFFu) << 16);               // tensor_dim1[15:0]
  g1[3] = (int)((tile_h >> 16) & 0xFFFFu)                // tensor_dim1[31:16]
        | (int)((tile_w & 0xFFFFu) << 16);               // tile_dim0
  g1[4] = (int)(tile_h & 0xFFFFu);                       // tile_dim1 (tile_dim2=0)
  g1[5] = (int)stride;                                   // dim0_stride[31:0]
  g1[6] = 0;                                             // stride[47:32]=0, dim1_stride lo
  g1[7] = 0;
  i32x4 z4 = {0, 0, 0, 0};
  i32x8 z8 = {0, 0, 0, 0, 0, 0, 0, 0};
  __builtin_amdgcn_tensor_load_to_lds(g0, g1, z4, z4, z8, 0);
#else
  (void)lds_addr; (void)gptr; (void)tile_w; (void)tile_h;
  (void)stride; (void)pad_int_code; (void)pad_dw;
#endif
}

// ---------------------------------------------------------------------------
// Elementwise f32 -> bf16 convert
// ---------------------------------------------------------------------------
__global__ __launch_bounds__(256)
void cvt_f32_bf16(const float* __restrict__ in, __bf16* __restrict__ out, int n) {
  int i0 = (blockIdx.x * 256 + threadIdx.x) * 4;
  if (i0 + 3 < n) {
    float4 v = *(const float4*)(in + i0);
    out[i0 + 0] = (__bf16)v.x;
    out[i0 + 1] = (__bf16)v.y;
    out[i0 + 2] = (__bf16)v.z;
    out[i0 + 3] = (__bf16)v.w;
  } else {
    for (int i = i0; i < n; ++i) out[i] = (__bf16)in[i];
  }
}

// ---------------------------------------------------------------------------
// W (N x K f32, row-major) -> Wt (K x N bf16, row-major)   [32x32 LDS tiles]
// ---------------------------------------------------------------------------
__global__ __launch_bounds__(256)
void transpose_cvt(const float* __restrict__ W, __bf16* __restrict__ Wt, int N, int K) {
  __shared__ float tile[32][33];
  int k0 = blockIdx.x * 32, n0 = blockIdx.y * 32;
  int tx = threadIdx.x, ty = threadIdx.y;            // 32 x 8
  for (int i = 0; i < 32; i += 8)
    tile[ty + i][tx] = W[(size_t)(n0 + ty + i) * K + k0 + tx];
  __syncthreads();
  for (int i = 0; i < 32; i += 8)
    Wt[(size_t)(k0 + ty + i) * N + n0 + tx] = (__bf16)tile[tx][ty + i];
}

// ---------------------------------------------------------------------------
// GEMM: Y[M,N] = alpha * A[M,K] @ Bt[K,N]   (A,Bt bf16 row-major)
// Templated on output mode so only one epilogue path is live (no acc spill).
// Block 256 thr (8 waves), tile 128x128, wave tile 32x64, K-step 32,
// double-buffered LDS. TDM path: wave 0 DMAs next tiles while all compute.
// ---------------------------------------------------------------------------
#define A_PITCH 80     // 128 rows x (64B data + 16B pad)
#define B_PITCH 272    // 32 rows  x (256B data + 16B pad)
#define LDS_BUF (128 * A_PITCH + 32 * B_PITCH)

template <bool WF, bool WH>
__global__ __launch_bounds__(256)
void gemm_bf16(const __bf16* __restrict__ A, const __bf16* __restrict__ Bt,
               float* __restrict__ outF, __bf16* __restrict__ outH,
               int M, int N, int K, float alpha) {
  const int tid  = threadIdx.x;
  const int lane = tid & 31;
  const int wid  = tid >> 5;
  const int wm   = wid & 3;      // wave row   (x32)
  const int wn   = wid >> 2;     // wave col   (x64)
  const int half = lane >> 4;
  const int ln   = lane & 15;
  const int m0 = blockIdx.y * 128, n0 = blockIdx.x * 128;

  __shared__ __align__(16) unsigned char smem[2 * LDS_BUF];

  v8f acc[2][4];
  for (int i = 0; i < 2; ++i)
    for (int j = 0; j < 4; ++j) acc[i][j] = zero8();

  auto compute = [&](int buf) {
    const unsigned char* as = smem + buf * LDS_BUF;
    const unsigned char* bs = as + 128 * A_PITCH;
    // A frags: lane<16 -> M=ln, K bytes {0..15,32..47}; lane>=16 -> {16..31,48..63}
    BF16x16 fa[2];
    const int c0 = half * 16;
    for (int mi = 0; mi < 2; ++mi) {
      const unsigned char* p = as + (wm * 32 + mi * 16 + ln) * A_PITCH + c0;
      fa[mi].h[0] = *(const v8bf*)(p);
      fa[mi].h[1] = *(const v8bf*)(p + 32);
    }
    // B frags: lane -> N=ln, K = {half*8..+7, 16+half*8..+7} gathered down k-rows
    BF16x16 fb[4];
    for (int ni = 0; ni < 4; ++ni) {
      const unsigned char* p = bs + (wn * 64 + ni * 16 + ln) * 2;
      for (int s = 0; s < 16; ++s)
        fb[ni].v[s] = *(const __bf16*)(p + ((s & 8) * 2 + half * 8 + (s & 7)) * B_PITCH);
    }
    for (int mi = 0; mi < 2; ++mi)
      for (int ni = 0; ni < 4; ++ni)
        acc[mi][ni] = wmma_bf16(fa[mi].v, fb[ni].v, acc[mi][ni]);
  };

  const int nk = K / 32;

#if USE_TDM
  auto issue = [&](int buf, int kt) {
    unsigned char* base = smem + buf * LDS_BUF;
    tdm_load_2d((unsigned)(uintptr_t)base,
                A + (size_t)m0 * K + kt * 32, 32, 128, (unsigned)K, 3, 4);
    tdm_load_2d((unsigned)(uintptr_t)(base + 128 * A_PITCH),
                Bt + (size_t)kt * 32 * N + n0, 128, 32, (unsigned)N, 5, 4);
  };
  if (wid == 0) {
    issue(0, 0);
    __builtin_amdgcn_s_wait_tensorcnt(0);
  }
  __syncthreads();
  for (int kt = 0; kt < nk; ++kt) {
    const int buf = kt & 1;
    if (wid == 0 && kt + 1 < nk) issue(buf ^ 1, kt + 1);  // DMA next, overlap
    compute(buf);
    if (wid == 0 && kt + 1 < nk) __builtin_amdgcn_s_wait_tensorcnt(0);
    __syncthreads();   // publishes TDM completion to all waves
  }
#else
  v8bf ra0, ra1, rb0, rb1;
  auto loadg = [&](int kt) {
    const int k0 = kt * 32;
    const __bf16* ap = A + (size_t)(m0 + (tid >> 2)) * K + k0 + (tid & 3) * 8;
    ra0 = *(const v8bf*)ap;
    ra1 = *(const v8bf*)(ap + (size_t)64 * K);
    const __bf16* bp = Bt + (size_t)(k0 + (tid >> 4)) * N + n0 + (tid & 15) * 8;
    rb0 = *(const v8bf*)bp;
    rb1 = *(const v8bf*)(bp + (size_t)16 * N);
  };
  auto stlds = [&](int buf) {
    unsigned char* as = smem + buf * LDS_BUF;
    unsigned char* bs = as + 128 * A_PITCH;
    *(v8bf*)(as + (tid >> 2) * A_PITCH + (tid & 3) * 16)          = ra0;
    *(v8bf*)(as + ((tid >> 2) + 64) * A_PITCH + (tid & 3) * 16)   = ra1;
    *(v8bf*)(bs + (tid >> 4) * B_PITCH + (tid & 15) * 16)         = rb0;
    *(v8bf*)(bs + ((tid >> 4) + 16) * B_PITCH + (tid & 15) * 16)  = rb1;
  };
  loadg(0);
  stlds(0);
  __syncthreads();
  for (int kt = 0; kt < nk; ++kt) {
    const int buf = kt & 1;
    if (kt + 1 < nk) loadg(kt + 1);
    if (kt + 2 < nk) {
      __builtin_prefetch(A + (size_t)(m0 + (tid >> 2)) * K + (kt + 2) * 32, 0, 1);
      __builtin_prefetch(Bt + (size_t)((kt + 2) * 32 + (tid >> 4)) * N + n0, 0, 1);
    }
    compute(buf);
    if (kt + 1 < nk) stlds(buf ^ 1);
    __syncthreads();
  }
#endif

  // Epilogue (single path, templated). C layout: vgpr r -> row r / r+8, col = ln.
  for (int mi = 0; mi < 2; ++mi)
    for (int ni = 0; ni < 4; ++ni)
      for (int r = 0; r < 8; ++r) {
        const int row = m0 + wm * 32 + mi * 16 + r + half * 8;
        const int col = n0 + wn * 64 + ni * 16 + ln;
        const float v = acc[mi][ni][r] * alpha;
        if (WF) outF[(size_t)row * N + col] = v;
        if (WH) outH[(size_t)row * N + col] = (__bf16)v;
      }
}

// ---------------------------------------------------------------------------
// Scatter new K rows: Y (b,s,h*dh) f32 -> d_out K (bh, 2048+q, d) f32
//                                      -> Kt ws   (bh, d, 2048+q) bf16 (d-major)
// ---------------------------------------------------------------------------
__global__ __launch_bounds__(256)
void scatter_new_k(const float* __restrict__ Y, float* __restrict__ outK,
                   __bf16* __restrict__ Kt) {
  int e0 = (blockIdx.x * 256 + threadIdx.x) * 4;
  for (int u = 0; u < 4; ++u) {
    int e = e0 + u;
    int row = e >> 11, col = e & 2047;
    int b = row >> 11, q = row & 2047;
    int h = col >> 7,  d = col & 127;
    int bh = b * 16 + h, j = 2048 + q;
    float v = Y[(size_t)e];
    outK[((size_t)bh * 4096 + j) * 128 + d] = v;
    Kt[((size_t)bh * 128 + d) * 4096 + j] = (__bf16)v;
  }
}

__global__ __launch_bounds__(256)
void scatter_new_v(const float* __restrict__ Y, float* __restrict__ outV,
                   __bf16* __restrict__ Vh) {
  int e0 = (blockIdx.x * 256 + threadIdx.x) * 4;
  for (int u = 0; u < 4; ++u) {
    int e = e0 + u;
    int row = e >> 11, col = e & 2047;
    int b = row >> 11, q = row & 2047;
    int h = col >> 7,  d = col & 127;
    int bh = b * 16 + h, j = 2048 + q;
    float v = Y[(size_t)e];
    outV[((size_t)bh * 4096 + j) * 128 + d] = v;
    Vh[((size_t)bh * 4096 + j) * 128 + d] = (__bf16)v;
  }
}

__global__ __launch_bounds__(256)
void past_kv(const float* __restrict__ pk, const float* __restrict__ pv,
             float* __restrict__ outK, float* __restrict__ outV,
             __bf16* __restrict__ Kt, __bf16* __restrict__ Vh) {
  int e0 = (blockIdx.x * 256 + threadIdx.x) * 4;
  for (int u = 0; u < 4; ++u) {
    int e = e0 + u;
    int bh = e >> 18;
    int rem = e & 262143;
    int j = rem >> 7, d = rem & 127;
    float k = pk[(size_t)e];
    float v = pv[(size_t)e];
    outK[((size_t)bh * 4096 + j) * 128 + d] = k;
    outV[((size_t)bh * 4096 + j) * 128 + d] = v;
    Kt[((size_t)bh * 128 + d) * 4096 + j] = (__bf16)k;
    Vh[((size_t)bh * 4096 + j) * 128 + d] = (__bf16)v;
  }
}

// ---------------------------------------------------------------------------
// Flash attention. Grid (qtile=32, bh=32), block 128 (4 waves).
// Wave w handles 16 query rows; block shares 64-key K/V LDS tiles.
// Q pre-scaled by 1/sqrt(dh). Kt is d-major (bh,128,4096); Vh (bh,4096,128).
// TDM path double-buffers K/V tiles (wave-0 DMA overlapped with compute).
// ---------------------------------------------------------------------------
#define KS_PITCH 144   // 128B data + 16B pad, per d-row of 64 keys
#define VS_PITCH 272   // 256B data + 16B pad, per key-row of 128 d
#define PS_PITCH 144   // 16x64 bf16 P tile per wave
#define KS_BYTES (128 * KS_PITCH)
#define VS_BYTES (64 * VS_PITCH)
#define KV_BYTES (KS_BYTES + VS_BYTES)

__global__ __launch_bounds__(128)
void flash_attn(const __bf16* __restrict__ Qh, const __bf16* __restrict__ Kt,
                const __bf16* __restrict__ Vh, __bf16* __restrict__ Oh) {
  const int qt = blockIdx.x;
  const int bh = blockIdx.y;
  const int b = bh >> 4, h = bh & 15;
  const int tid = threadIdx.x, wid = tid >> 5, lane = tid & 31;
  const int half = lane >> 4, ln = lane & 15;

#if USE_TDM
  __shared__ __align__(16) unsigned char smem[2 * KV_BYTES + 4 * 16 * PS_PITCH];
  unsigned char* PsB = smem + 2 * KV_BYTES + wid * 16 * PS_PITCH;
#else
  __shared__ __align__(16) unsigned char smem[KV_BYTES + 4 * 16 * PS_PITCH];
  unsigned char* PsB = smem + KV_BYTES + wid * 16 * PS_PITCH;
#endif

  const int q0 = qt * 64 + wid * 16;

  // Q A-fragments for 4 d-chunks of 32, kept in registers for the whole loop
  BF16x16 fq[4];
  {
    const __bf16* qrow = Qh + ((size_t)(b * 2048 + q0 + ln)) * 2048 + h * 128;
    for (int c = 0; c < 4; ++c) {
      const __bf16* p = qrow + c * 32 + half * 8;
      fq[c].h[0] = *(const v8bf*)(p);
      fq[c].h[1] = *(const v8bf*)(p + 16);
    }
  }

  v8f o[8];
  for (int i = 0; i < 8; ++i) o[i] = zero8();
  float mst[8], lst[8];
  for (int r = 0; r < 8; ++r) { mst[r] = -1e30f; lst[r] = 0.f; }

  const __bf16* kbase = Kt + (size_t)bh * 128 * 4096;
  const __bf16* vbase = Vh + (size_t)bh * 4096 * 128;
  const int nj = 33 + qt;   // causal: keys [0, 2048 + q0 + 63]

#if USE_TDM
  auto issue = [&](int buf, int j0) {
    unsigned char* kb = smem + buf * KV_BYTES;
    tdm_load_2d((unsigned)(uintptr_t)kb, kbase + j0, 64, 128, 4096u, 4, 4);
    tdm_load_2d((unsigned)(uintptr_t)(kb + KS_BYTES),
                vbase + (size_t)j0 * 128, 128, 64, 128u, 5, 4);
  };
  if (wid == 0) {
    issue(0, 0);
    __builtin_amdgcn_s_wait_tensorcnt(0);
  }
  __syncthreads();
#endif

  for (int jt = 0; jt < nj; ++jt) {
    const int j0 = jt * 64;
#if USE_TDM
    const int buf = jt & 1;
    if (wid == 0 && jt + 1 < nj) issue(buf ^ 1, (jt + 1) * 64);
    const unsigned char* KsB = smem + buf * KV_BYTES;
    const unsigned char* VsB = KsB + KS_BYTES;
#else
    const unsigned char* KsB = smem;
    const unsigned char* VsB = smem + KS_BYTES;
    __syncthreads();
    {
      const __bf16* src = kbase + (size_t)tid * 4096 + j0;
      unsigned char* dst = (unsigned char*)smem + tid * KS_PITCH;
      for (int c = 0; c < 8; ++c)
        *(v8bf*)(dst + c * 16) = *(const v8bf*)(src + c * 8);
      const __bf16* vsrc = vbase + (size_t)(j0 + (tid >> 1)) * 128 + (tid & 1) * 64;
      unsigned char* vdst = (unsigned char*)smem + KS_BYTES +
                            (tid >> 1) * VS_PITCH + (tid & 1) * 128;
      for (int c = 0; c < 8; ++c)
        *(v8bf*)(vdst + c * 16) = *(const v8bf*)(vsrc + c * 8);
    }
    __syncthreads();
    if (jt + 1 < nj) {
      __builtin_prefetch(kbase + (size_t)tid * 4096 + j0 + 64, 0, 1);
      __builtin_prefetch(vbase + (size_t)(j0 + 64 + (tid >> 1)) * 128, 0, 1);
    }
#endif

    // S = Q @ K^T : 4 key sub-tiles of 16, reduce over 4 d-chunks of 32
    v8f s[4];
    for (int i = 0; i < 4; ++i) s[i] = zero8();
    for (int dc = 0; dc < 4; ++dc) {
      const unsigned char* kp = KsB + dc * 32 * KS_PITCH + ln * 2;
      for (int ni = 0; ni < 4; ++ni) {
        BF16x16 fb;
        for (int s = 0; s < 16; ++s)
          fb.v[s] = *(const __bf16*)(kp + ((s & 8) * 2 + half * 8 + (s & 7)) * KS_PITCH + ni * 32);
        s[ni] = wmma_bf16(fq[dc].v, fb.v, s[ni]);
      }
    }

    // causal mask + online softmax (per-row stats, rows r+8*half)
    float alpha[8];
    for (int r = 0; r < 8; ++r) {
      const int qrow = q0 + r + half * 8;
      const int lim = 2048 + qrow;
      float mx = mst[r];
      for (int ni = 0; ni < 4; ++ni) {
        const int jabs = j0 + ni * 16 + ln;
        if (jabs > lim) s[ni][r] = -1e30f;
        mx = fmaxf(mx, s[ni][r]);
      }
      for (int off = 8; off > 0; off >>= 1)
        mx = fmaxf(mx, __shfl_xor(mx, off, 16));
      alpha[r] = __expf(mst[r] - mx);
      float rs = 0.f;
      for (int ni = 0; ni < 4; ++ni) {
        float p = __expf(s[ni][r] - mx);
        s[ni][r] = p;
        rs += p;
      }
      for (int off = 8; off > 0; off >>= 1)
        rs += __shfl_xor(rs, off, 16);
      lst[r] = lst[r] * alpha[r] + rs;
      mst[r] = mx;
    }

    // rescale O, spill P to wave-private LDS, reload as A-fragments
    for (int di = 0; di < 8; ++di)
      for (int r = 0; r < 8; ++r) o[di][r] *= alpha[r];
    for (int ni = 0; ni < 4; ++ni)
      for (int r = 0; r < 8; ++r)
        *(__bf16*)(PsB + (r + 8 * half) * PS_PITCH + (ni * 16 + ln) * 2) =
            (__bf16)s[ni][r];
    BF16x16 fp[2];
    for (int c = 0; c < 2; ++c) {
      const unsigned char* p = PsB + ln * PS_PITCH + half * 16 + c * 64;
      fp[c].h[0] = *(const v8bf*)(p);
      fp[c].h[1] = *(const v8bf*)(p + 32);
    }

    // O += P @ V
    for (int kc = 0; kc < 2; ++kc) {
      const unsigned char* vp = VsB + kc * 32 * VS_PITCH + ln * 2;
      for (int di = 0; di < 8; ++di) {
        BF16x16 fv;
        for (int s = 0; s < 16; ++s)
          fv.v[s] = *(const __bf16*)(vp + ((s & 8) * 2 + half * 8 + (s & 7)) * VS_PITCH + di * 32);
        o[di] = wmma_bf16(fp[kc].v, fv.v, o[di]);
      }
    }

#if USE_TDM
    if (wid == 0 && jt + 1 < nj) __builtin_amdgcn_s_wait_tensorcnt(0);
    __syncthreads();
#endif
  }

  // normalize and store attn output (b, s, d_model) bf16
  for (int r = 0; r < 8; ++r) {
    const float inv = 1.0f / lst[r];
    const int qrow = q0 + r + half * 8;
    __bf16* dst = Oh + ((size_t)(b * 2048 + qrow)) * 2048 + h * 128 + ln;
    for (int di = 0; di < 8; ++di)
      dst[di * 16] = (__bf16)(o[di][r] * inv);
  }
}

// ---------------------------------------------------------------------------
// Host launcher
// ---------------------------------------------------------------------------
extern "C" void kernel_launch(void* const* d_in, const int* in_sizes, int n_in,
                              void* d_out, int out_size, void* d_ws, size_t ws_size,
                              hipStream_t stream) {
  const float* x      = (const float*)d_in[0];
  const float* past_k = (const float*)d_in[1];
  const float* past_v = (const float*)d_in[2];
  const float* Wq     = (const float*)d_in[3];
  const float* Wk     = (const float*)d_in[4];
  const float* Wv     = (const float*)d_in[5];
  const float* Wo     = (const float*)d_in[6];

  float* out  = (float*)d_out;           // (b, s, d_model) = 8,388,608
  float* outK = out + 8388608;           // (bh, 4096, 128) = 16,777,216
  float* outV = outK + 16777216;

  char* ws = (char*)d_ws;
  size_t off = 0;
  __bf16* xh    = (__bf16*)(ws + off); off += 16777216;   // x bf16
  __bf16* Wqt   = (__bf16*)(ws + off); off += 8388608;    // W^T bf16 (k-major)
  __bf16* Wkt   = (__bf16*)(ws + off); off += 8388608;
  __bf16* Wvt   = (__bf16*)(ws + off); off += 8388608;
  __bf16* Wot   = (__bf16*)(ws + off); off += 8388608;
  __bf16* Qh    = (__bf16*)(ws + off); off += 16777216;   // scaled Q, (b,s,dmodel)
  __bf16* Kth   = (__bf16*)(ws + off); off += 33554432;   // (bh, d, 4096)
  __bf16* Vhh   = (__bf16*)(ws + off); off += 33554432;   // (bh, 4096, d)
  __bf16* attnh = (__bf16*)(ws + off); off += 16777216;   // attn out bf16
  float*  Yf    = (float*)(ws + off);  off += 33554432;   // f32 GEMM scratch

  const int M = 4096, N = 2048, K = 2048;
  const float qscale = 0.08838834764831845f;   // 1/sqrt(128)

  cvt_f32_bf16<<<8192, 256, 0, stream>>>(x, xh, 8388608);

  dim3 tb(32, 8);
  dim3 tg(64, 64);
  transpose_cvt<<<tg, tb, 0, stream>>>(Wq, Wqt, N, K);
  transpose_cvt<<<tg, tb, 0, stream>>>(Wk, Wkt, N, K);
  transpose_cvt<<<tg, tb, 0, stream>>>(Wv, Wvt, N, K);
  transpose_cvt<<<tg, tb, 0, stream>>>(Wo, Wot, N, K);

  dim3 gg(N / 128, M / 128);   // (16, 32)

  // Q projection (bf16, pre-scaled)
  gemm_bf16<false, true><<<gg, 256, 0, stream>>>(xh, Wqt, (float*)nullptr, Qh,
                                                 M, N, K, qscale);
  // K projection -> f32 scratch, then scatter to d_out + d-major bf16
  gemm_bf16<true, false><<<gg, 256, 0, stream>>>(xh, Wkt, Yf, (__bf16*)nullptr,
                                                 M, N, K, 1.0f);
  scatter_new_k<<<8192, 256, 0, stream>>>(Yf, outK, Kth);
  // V projection (reuses Yf after scatter consumed it)
  gemm_bf16<true, false><<<gg, 256, 0, stream>>>(xh, Wvt, Yf, (__bf16*)nullptr,
                                                 M, N, K, 1.0f);
  scatter_new_v<<<8192, 256, 0, stream>>>(Yf, outV, Vhh);
  // past K/V -> outputs + bf16 working copies
  past_kv<<<8192, 256, 0, stream>>>(past_k, past_v, outK, outV, Kth, Vhh);

  // flash attention
  dim3 ag(32, 32);
  flash_attn<<<ag, 128, 0, stream>>>(Qh, Kth, Vhh, attnh);

  // output projection
  gemm_bf16<true, false><<<gg, 256, 0, stream>>>(attnh, Wot, out,
                                                 (__bf16*)nullptr, M, N, K, 1.0f);

  (void)in_sizes; (void)n_in; (void)out_size; (void)ws_size;
}
